// AttentionFusionModel_2937757630479
// MI455X (gfx1250) — matrix operand order repaired and checked by reading, not compiled, not run on hardware
//
#include <hip/hip_runtime.h>
#include <hip/hip_bf16.h>

// ---------------------------------------------------------------------------
// Fused dual-modal self-attention for MI455X (gfx1250), wave32 + WMMA.
//   D=200 padded to DP=224 (7 K-chunks of 32 for v_wmma_f32_16x16x32_f16).
//   Pipeline: weight-cvt -> QKV proj (f16) -> flash attention (async-LDS
//   double-buffered) -> out proj.
// ---------------------------------------------------------------------------

#define D0   200
#define DP   224           // padded D
#define SEQ  4096
#define NB   4
#define R1   (NB * SEQ)    // rows per modality  = 16384
#define RT   (2 * R1)      // total rows (both modalities) = 32768
#define NKT  (SEQ / 64)    // 64 key tiles

typedef __attribute__((ext_vector_type(16))) _Float16 v16h;
typedef __attribute__((ext_vector_type(8)))  float    v8f;

union V16HU { v16h v; unsigned int u[8]; };

__device__ __attribute__((always_inline)) inline v8f wmma16(v16h a, v16h b, v8f c) {
  // D = A(16x32 f16) * B(32x16 f16) + C(16x16 f32)
  return __builtin_amdgcn_wmma_f32_16x16x32_f16(false, a, false, b, (short)0, c,
                                                false, false);
}

// A-fragment (16x32, MxK) from row-major tile t[row][k], stride in halfs.
// ISA 7.12.2: lane m=lane&15 holds row m; half-wave group selects K sub-chunk.
__device__ __attribute__((always_inline)) inline
v16h frag_a(const _Float16* t, int stride, int row0, int k0) {
  const int lane = threadIdx.x & 31;
  const unsigned int* p = (const unsigned int*)
      (t + (size_t)(row0 + (lane & 15)) * stride + k0 + ((lane >> 4) << 3));
  V16HU r;
  r.u[0] = p[0]; r.u[1] = p[1]; r.u[2] = p[2];  r.u[3] = p[3];   // K 0..7  (+8g)
  r.u[4] = p[8]; r.u[5] = p[9]; r.u[6] = p[10]; r.u[7] = p[11];  // K 16..23(+8g)
  return r.v;
}

// B-fragment (32x16, KxN) for C = A * T^T where T is row-major t[n][k].
// B[k][n] = T[n0+n][k0+k]; lane n=lane&15, group selects K 8-chunks (K 0..7,16..23 +8g).
__device__ __attribute__((always_inline)) inline
v16h frag_bt(const _Float16* t, int stride, int n0, int k0) {
  const int lane = threadIdx.x & 31;
  const unsigned int* p = (const unsigned int*)
      (t + (size_t)(n0 + (lane & 15)) * stride + k0 + ((lane >> 4) << 3));
  V16HU r;
#pragma unroll
  for (int i = 0; i < 8; ++i) r.u[i] = p[i + (i & 4)];
  return r.v;
}

// Async global->LDS 16B copy (CDNA5 TDM-adjacent path, tracked by ASYNCcnt).
// %0 = per-lane LDS byte offset (VDST), %1 = per-lane 64-bit global address.
__device__ __attribute__((always_inline)) inline
void async_b128(unsigned lds_byte_off, const void* gptr) {
  asm volatile("global_load_async_to_lds_b128 %0, %1, off"
               : : "v"(lds_byte_off), "v"(gptr) : "memory");
}

// ---------------------------------------------------------------------------
// Kernel 0: convert + zero-pad the 4 weight matrices to f16 [4][DP][DP].
// ---------------------------------------------------------------------------
__global__ __launch_bounds__(256) void wcvt_kernel(
    const float* __restrict__ Wq, const float* __restrict__ Wk,
    const float* __restrict__ Wv, const float* __restrict__ Wo,
    _Float16* __restrict__ Wf) {
  int idx = blockIdx.x * 256 + threadIdx.x;
  if (idx >= 4 * DP * DP) return;
  int w   = idx / (DP * DP);
  int rem = idx - w * (DP * DP);
  int r = rem / DP, c = rem - r * DP;
  const float* src = (w == 0) ? Wq : (w == 1) ? Wk : (w == 2) ? Wv : Wo;
  Wf[idx] = (r < D0 && c < D0) ? (_Float16)src[r * D0 + c] : (_Float16)0.0f;
}

// ---------------------------------------------------------------------------
// Kernel 1: QKV projection for both modalities (rows 0..RT-1).
//   Q,K stored row-major f16 [RT][DP]; V stored transposed f16 [8][DP][SEQ].
//   Q pre-scaled by 1/sqrt(D).
// ---------------------------------------------------------------------------
__global__ __launch_bounds__(128) void qkv_kernel(
    const float* __restrict__ m1, const float* __restrict__ m2,
    const float* __restrict__ bq, const float* __restrict__ bk,
    const float* __restrict__ bv, const _Float16* __restrict__ Wf,
    _Float16* __restrict__ Qg, _Float16* __restrict__ Kg,
    _Float16* __restrict__ Vtg) {
  __shared__ _Float16 Xs[64 * DP];
  const int tid  = threadIdx.x;
  const int row0 = blockIdx.x * 64;           // global row (modalities stacked)
  const int bm   = row0 >> 12;                // combined (mod,batch) 0..7
  const float* X = (row0 < R1) ? (m1 + (size_t)row0 * D0)
                               : (m2 + (size_t)(row0 - R1) * D0);
  for (int i = tid; i < 64 * DP; i += 128) {
    int r = i / DP, c = i - r * DP;
    Xs[i] = (c < D0) ? (_Float16)X[(size_t)r * D0 + c] : (_Float16)0.0f;
  }
  __syncthreads();

  const int wid = tid >> 5, lane = tid & 31;
  const int n = lane & 15, g = lane >> 4;

  for (int w = 0; w < 3; ++w) {
    const _Float16* W  = Wf + (size_t)w * DP * DP;
    const float* bias  = (w == 0) ? bq : ((w == 1) ? bk : bv);
    const float  scale = (w == 0) ? 0.07071067811865475f : 1.0f;  // 1/sqrt(200)
    for (int h = 0; h < 2; ++h) {             // two N-halves to cap VGPR use
      const int nbase = h * 112;
      v8f acc[7];
#pragma unroll
      for (int nt = 0; nt < 7; ++nt) {
        int   j  = nbase + nt * 16 + n;
        float b0 = (j < D0) ? bias[j] : 0.0f;
#pragma unroll
        for (int p = 0; p < 8; ++p) acc[nt][p] = b0;
      }
      for (int c = 0; c < 7; ++c) {
        v16h a = frag_a(Xs, DP, wid * 16, c * 32);
#pragma unroll
        for (int nt = 0; nt < 7; ++nt)
          acc[nt] = wmma16(a, frag_bt(W, DP, nbase + nt * 16, c * 32), acc[nt]);
      }
#pragma unroll
      for (int nt = 0; nt < 7; ++nt) {
        const int col = nbase + nt * 16 + n;
#pragma unroll
        for (int p = 0; p < 8; ++p) {
          const int row = row0 + wid * 16 + p + 8 * g;
          const _Float16 val = (_Float16)(acc[nt][p] * scale);
          if (w == 0)      Qg[(size_t)row * DP + col] = val;
          else if (w == 1) Kg[(size_t)row * DP + col] = val;
          else  Vtg[((size_t)bm * DP + col) * SEQ + (row & (SEQ - 1))] = val;
        }
      }
    }
  }
}

// ---------------------------------------------------------------------------
// Kernel 2: flash attention with async-LDS double buffering.
//   LDS (halfs): [K0][K1][V0][V1][Q][P] ; each tile buffer = 64*DP = 14336.
//   Tile kt lives in buffer kt&1. While computing tile kt we async-load
//   tile kt+1 (28 async ops/wave), then s_wait_asynccnt 28 retires tile kt.
// ---------------------------------------------------------------------------
#define TILEH (64 * DP)                     // halfs per tile buffer (=14336)
#define ATTN_LDS_BYTES ((5 * TILEH + 4 * 16 * 64) * 2)   // 151,552 B

__device__ __attribute__((always_inline)) inline
void stage_kv_async(const _Float16* Kb, const _Float16* Vtb, int kt,
                    unsigned koff, unsigned voff, int tid) {
  const _Float16* ksrc = Kb + (size_t)kt * 64 * DP;
#pragma unroll 1
  for (int i = tid; i < 1792; i += 128)          // 64 x DP halfs, 16B chunks
    async_b128(koff + (unsigned)i * 16u, ksrc + (size_t)i * 8);
#pragma unroll 1
  for (int i = tid; i < 1792; i += 128) {        // DP rows x 64 keys
    int r = i >> 3, c = i & 7;
    async_b128(voff + (unsigned)i * 16u,
               Vtb + (size_t)r * SEQ + kt * 64 + c * 8);
  }
}

__global__ __launch_bounds__(128) void attn_kernel(
    const _Float16* __restrict__ Qg, const _Float16* __restrict__ Kg,
    const _Float16* __restrict__ Vtg, _Float16* __restrict__ Ag) {
  extern __shared__ _Float16 smem[];
  const unsigned OFF_K0 = 0;
  const unsigned OFF_K1 = TILEH * 2u;
  const unsigned OFF_V0 = TILEH * 4u;
  const unsigned OFF_V1 = TILEH * 6u;
  const unsigned OFF_Q  = TILEH * 8u;           // byte offsets
  _Float16* Qs = smem + 4 * TILEH;
  _Float16* Ps = smem + 5 * TILEH;              // [4 waves][16][64]

  const int tid = threadIdx.x, wid = tid >> 5, lane = tid & 31;
  const int n = lane & 15, g = lane >> 4;
  const int bm = blockIdx.x >> 6;               // 0..7
  const int qt = blockIdx.x & 63;
  const size_t qrow0 = (size_t)bm * SEQ + qt * 64;

  const _Float16* Kb  = Kg  + (size_t)bm * SEQ * DP;
  const _Float16* Vtb = Vtg + (size_t)bm * DP * SEQ;
  _Float16* Pw = Ps + wid * (16 * 64);

  // Prologue: async-stage Q tile (14 ops/wave) and K/V tile 0 (28 ops/wave).
  {
    const _Float16* qsrc = Qg + qrow0 * DP;
#pragma unroll 1
    for (int i = tid; i < 1792; i += 128)
      async_b128(OFF_Q + (unsigned)i * 16u, qsrc + (size_t)i * 8);
  }
  stage_kv_async(Kb, Vtb, 0, OFF_K0, OFF_V0, tid);

  float mrow[8], lrow[8];
  v8f acc[14];
#pragma unroll
  for (int p = 0; p < 8; ++p) { mrow[p] = -1e30f; lrow[p] = 0.0f; }
#pragma unroll
  for (int nt = 0; nt < 14; ++nt)
#pragma unroll
    for (int p = 0; p < 8; ++p) acc[nt][p] = 0.0f;

  for (int kt = 0; kt < NKT; ++kt) {
    const int cur = kt & 1;
    __syncthreads();   // all waves done reading buffer cur^1 from iter kt-1
    if (kt + 1 < NKT) {
      stage_kv_async(Kb, Vtb, kt + 1,
                     cur ? OFF_K0 : OFF_K1, cur ? OFF_V0 : OFF_V1, tid);
      // Retire everything except the newest 28 ops (= tile kt+1): async loads
      // complete in order, so tile kt (and the Q prologue) are now in LDS.
      asm volatile("s_wait_asynccnt 28" ::: "memory");
    } else {
      asm volatile("s_wait_asynccnt 0" ::: "memory");
    }
    __syncthreads();   // tile kt visible to all waves

    const _Float16* Ks = smem + (cur ? TILEH : 0);
    const _Float16* Vs = smem + (cur ? 3 * TILEH : 2 * TILEH);

    // ---- scores S = Q * K^T  (16 rows x 64 keys, fp32) ----
    v8f s[4];
#pragma unroll
    for (int j = 0; j < 4; ++j)
#pragma unroll
      for (int p = 0; p < 8; ++p) s[j][p] = 0.0f;
    for (int c = 0; c < 7; ++c) {
      v16h a = frag_a(Qs, DP, wid * 16, c * 32);
#pragma unroll
      for (int j = 0; j < 4; ++j)
        s[j] = wmma16(a, frag_bt(Ks, DP, j * 16, c * 32), s[j]);
    }

    // ---- online softmax (row reductions across the 16-lane N-group) ----
    float alpha[8], rsum[8];
#pragma unroll
    for (int p = 0; p < 8; ++p) {
      float t = fmaxf(fmaxf(s[0][p], s[1][p]), fmaxf(s[2][p], s[3][p]));
      t = fmaxf(t, __shfl_xor(t, 1, 32));
      t = fmaxf(t, __shfl_xor(t, 2, 32));
      t = fmaxf(t, __shfl_xor(t, 4, 32));
      t = fmaxf(t, __shfl_xor(t, 8, 32));
      float mnew = fmaxf(mrow[p], t);
      alpha[p] = __expf(mrow[p] - mnew);
      mrow[p]  = mnew;
      rsum[p]  = 0.0f;
    }
#pragma unroll
    for (int j = 0; j < 4; ++j)
#pragma unroll
      for (int p = 0; p < 8; ++p) {
        float e = __expf(s[j][p] - mrow[p]);
        rsum[p] += e;
        Pw[(p + 8 * g) * 64 + j * 16 + n] = (_Float16)e;  // C-layout -> row-major
      }
#pragma unroll
    for (int p = 0; p < 8; ++p) {
      float t = rsum[p];
      t += __shfl_xor(t, 1, 32);
      t += __shfl_xor(t, 2, 32);
      t += __shfl_xor(t, 4, 32);
      t += __shfl_xor(t, 8, 32);
      lrow[p] = lrow[p] * alpha[p] + t;
    }
#pragma unroll
    for (int nt = 0; nt < 14; ++nt)
#pragma unroll
      for (int p = 0; p < 8; ++p) acc[nt][p] *= alpha[p];

    // ---- acc += P(16x64) * V(64xDP)  via Vt tile: out = P * Vs^T ----
    v16h a0 = frag_a(Pw, 64, 0, 0);
    v16h a1 = frag_a(Pw, 64, 0, 32);
#pragma unroll
    for (int nt = 0; nt < 14; ++nt) {
      acc[nt] = wmma16(a0, frag_bt(Vs, 64, nt * 16, 0),  acc[nt]);
      acc[nt] = wmma16(a1, frag_bt(Vs, 64, nt * 16, 32), acc[nt]);
    }
  }

  // ---- finalize: divide by l, store f16 attention output row-major ----
  float inv[8];
#pragma unroll
  for (int p = 0; p < 8; ++p) inv[p] = 1.0f / lrow[p];
#pragma unroll
  for (int nt = 0; nt < 14; ++nt)
#pragma unroll
    for (int p = 0; p < 8; ++p) {
      size_t row = qrow0 + wid * 16 + p + 8 * g;
      Ag[row * DP + nt * 16 + n] = (_Float16)(acc[nt][p] * inv[p]);
    }
}

// ---------------------------------------------------------------------------
// Kernel 3: fused = a1 + a2, then out = fused @ Wo^T + bo  (fp32 output).
// ---------------------------------------------------------------------------
__global__ __launch_bounds__(128) void outp_kernel(
    const _Float16* __restrict__ Ag, const _Float16* __restrict__ Wf,
    const float* __restrict__ bo, float* __restrict__ out) {
  __shared__ _Float16 Fs[64 * DP];
  const int tid  = threadIdx.x;
  const int row0 = blockIdx.x * 64;            // 0..R1-1
  const _Float16* A1 = Ag + (size_t)row0 * DP;
  const _Float16* A2 = Ag + (size_t)(row0 + R1) * DP;
  for (int i = tid; i < 64 * DP; i += 128)
    Fs[i] = (_Float16)((float)A1[i] + (float)A2[i]);
  __syncthreads();

  const int wid = tid >> 5, lane = tid & 31;
  const int n = lane & 15, g = lane >> 4;
  const _Float16* W = Wf + (size_t)3 * DP * DP;   // Wo

  for (int h = 0; h < 2; ++h) {
    const int nbase = h * 112;
    v8f acc[7];
#pragma unroll
    for (int nt = 0; nt < 7; ++nt) {
      int   j  = nbase + nt * 16 + n;
      float b0 = (j < D0) ? bo[j] : 0.0f;
#pragma unroll
      for (int p = 0; p < 8; ++p) acc[nt][p] = b0;
    }
    for (int c = 0; c < 7; ++c) {
      v16h a = frag_a(Fs, DP, wid * 16, c * 32);
#pragma unroll
      for (int nt = 0; nt < 7; ++nt)
        acc[nt] = wmma16(a, frag_bt(W, DP, nbase + nt * 16, c * 32), acc[nt]);
    }
#pragma unroll
    for (int nt = 0; nt < 7; ++nt) {
      const int col = nbase + nt * 16 + n;
      if (col < D0) {
#pragma unroll
        for (int p = 0; p < 8; ++p) {
          const int row = row0 + wid * 16 + p + 8 * g;
          out[(size_t)row * D0 + col] = acc[nt][p];
        }
      }
    }
  }
}

// ---------------------------------------------------------------------------
// Launch. Workspace layout (bytes):
//   Wf16 [4][DP][DP]          @ 0          (401,408)
//   Q    [RT][DP] f16         @ 401,408    (14,680,064)
//   K    [RT][DP] f16         @ 15,081,472 (14,680,064)
//   Vt   [8][DP][SEQ] f16     @ 29,761,536 (14,680,064)
//   A    [RT][DP] f16         @ 44,441,600 (14,680,064)   total ~59.2 MB
// ---------------------------------------------------------------------------
extern "C" void kernel_launch(void* const* d_in, const int* in_sizes, int n_in,
                              void* d_out, int out_size, void* d_ws,
                              size_t ws_size, hipStream_t stream) {
  const float* m1 = (const float*)d_in[0];
  const float* m2 = (const float*)d_in[1];
  const float* Wq = (const float*)d_in[2];
  const float* bq = (const float*)d_in[3];
  const float* Wk = (const float*)d_in[4];
  const float* bk = (const float*)d_in[5];
  const float* Wv = (const float*)d_in[6];
  const float* bv = (const float*)d_in[7];
  const float* Wo = (const float*)d_in[8];
  const float* bo = (const float*)d_in[9];

  char* ws = (char*)d_ws;
  const size_t SZW = (size_t)4 * DP * DP * 2;
  const size_t SZM = (size_t)RT * DP * 2;
  _Float16* Wf  = (_Float16*)(ws);
  _Float16* Qg  = (_Float16*)(ws + SZW);
  _Float16* Kg  = (_Float16*)(ws + SZW + SZM);
  _Float16* Vtg = (_Float16*)(ws + SZW + 2 * SZM);
  _Float16* Ag  = (_Float16*)(ws + SZW + 3 * SZM);

  wcvt_kernel<<<(4 * DP * DP + 255) / 256, 256, 0, stream>>>(Wq, Wk, Wv, Wo, Wf);
  qkv_kernel<<<RT / 64, 128, 0, stream>>>(m1, m2, bq, bk, bv, Wf, Qg, Kg, Vtg);

  // 148KB dynamic LDS -> 2 workgroups per 320KB WGP.
  attn_kernel<<<8 * NKT, 128, ATTN_LDS_BYTES, stream>>>(Qg, Kg, Vtg, Ag);

  outp_kernel<<<R1 / 64, 128, 0, stream>>>(Ag, Wf, bo, (float*)d_out);
}
